// TorchAnalysisGNN_87033217286449
// MI455X (gfx1250) — compile-verified, matrix operand order checked by reading, not run on hardware
//
#include <hip/hip_runtime.h>

// ---------------------------------------------------------------------------
// Types for WMMA (CDNA5 / gfx1250, wave32)
// ---------------------------------------------------------------------------
typedef __attribute__((ext_vector_type(16))) __bf16 v16bf;
typedef __attribute__((ext_vector_type(8)))  float  v8f;
typedef __attribute__((ext_vector_type(4)))  unsigned int u32x4;

union Frag {
    u32x4 q[2];
    v16bf v;
};

__device__ __forceinline__ unsigned short f2bf(float f) {
    union { float f; unsigned u; } x; x.f = f;
    unsigned r = x.u + 0x7FFFu + ((x.u >> 16) & 1u);
    return (unsigned short)(r >> 16);
}
__device__ __forceinline__ float bf2f(unsigned short b) {
    union { unsigned u; float f; } x; x.u = ((unsigned)b) << 16;
    return x.f;
}

// ---------------------------------------------------------------------------
// Weight packing: fp32 [K][N] (or transposed [N][K]) -> bf16 packed into the
// per-lane WMMA B-fragment layout.  For each (ntile of 16 cols, ktile of 32 k):
// 32 lanes x 16 bf16 contiguous.  Element (lane, e) holds
//   k = ktile*32 + (e>>3)*16 + (lane>>4)*8 + (e&7),  n = ntile*16 + (lane&15)
// matching the 16-bit B operand layout of V_WMMA_*_16X16X32.
// ---------------------------------------------------------------------------
__global__ __launch_bounds__(256) void pack_weight_k(
    const float* __restrict__ W, unsigned short* __restrict__ P,
    int Kact, int Nact, int Kpad, int Npad, int transposed)
{
    long tid = (long)blockIdx.x * 256 + threadIdx.x;
    long total = (long)(Npad >> 4) * (Kpad >> 5) * 512;
    if (tid >= total) return;
    int  e    = (int)(tid & 15);
    int  lane = (int)((tid >> 4) & 31);
    long blk  = tid >> 9;
    int  KT   = Kpad >> 5;
    int  ktile = (int)(blk % KT);
    int  ntile = (int)(blk / KT);
    int k = ktile * 32 + ((e >> 3) << 4) + ((lane >> 4) << 3) + (e & 7);
    int n = ntile * 16 + (lane & 15);
    float v = 0.f;
    if (k < Kact && n < Nact)
        v = transposed ? W[(long)n * Kact + k] : W[(long)k * Nact + n];
    P[tid] = f2bf(v);
}

// ---------------------------------------------------------------------------
// Inner K-loop of the WMMA GEMM, manually unswitched on whether the second
// 16-col tile is valid (T1).  Both specializations are straight-line
// load+WMMA loops with full EXEC (ISA requirement for WMMA) and no
// per-iteration mask juggling, so the compiler can unroll & pipeline them.
// ---------------------------------------------------------------------------
template <bool T1>
__device__ __forceinline__ void gemm_core(
    const unsigned short* __restrict__ arow0,
    const unsigned short* __restrict__ arow1,
    const unsigned short* __restrict__ bcol0,
    const unsigned short* __restrict__ bcol1,
    int Kin, v8f& acc00, v8f& acc01, v8f& acc10, v8f& acc11)
{
    for (int kt = 0; kt < Kin; kt += 32) {
        Frag a0, a1, b0, b1;
        a0.q[0] = *(const u32x4*)(arow0 + kt);
        a0.q[1] = *(const u32x4*)(arow0 + kt + 16);
        a1.q[0] = *(const u32x4*)(arow1 + kt);
        a1.q[1] = *(const u32x4*)(arow1 + kt + 16);
        long bo = (long)(kt >> 5) << 9;
        b0.q[0] = *(const u32x4*)(bcol0 + bo);
        b0.q[1] = *(const u32x4*)(bcol0 + bo + 8);
        if (T1) {
            b1.q[0] = *(const u32x4*)(bcol1 + bo);
            b1.q[1] = *(const u32x4*)(bcol1 + bo + 8);
        }
        acc00 = __builtin_amdgcn_wmma_f32_16x16x32_bf16(
            false, a0.v, false, b0.v, (short)0, acc00, false, false);
        acc10 = __builtin_amdgcn_wmma_f32_16x16x32_bf16(
            false, a1.v, false, b0.v, (short)0, acc10, false, false);
        if (T1) {
            acc01 = __builtin_amdgcn_wmma_f32_16x16x32_bf16(
                false, a0.v, false, b1.v, (short)0, acc01, false, false);
            acc11 = __builtin_amdgcn_wmma_f32_16x16x32_bf16(
                false, a1.v, false, b1.v, (short)0, acc11, false, false);
        }
    }
}

// ---------------------------------------------------------------------------
// Generic bf16 WMMA GEMM:  C[M, Nout] = act(A[M,Kin] * B + bias)
// 256 threads = 8 waves as 4 row-groups x 2 col-groups.
// Each wave computes a 32x32 tile = 2x2 WMMA fragments, reusing each A
// fragment across 2 B tiles and vice versa: 8 b128 loads per 4 WMMAs
// (1 KB of operand traffic per 16.4 KFLOP WMMA instead of 2 KB).
// Workgroup tile: 128 rows x 64 cols.  A: bf16 row-major (lda elems, rows
// padded to a multiple of 128).  Bp: packed layout above; per-16-col-tile
// validity guard (ntiles) supports N padded to 48 for the NC=38 heads.
// ---------------------------------------------------------------------------
__global__ __launch_bounds__(256) void gemm_bf16_wmma(
    const unsigned short* __restrict__ A, int lda,
    const unsigned short* __restrict__ Bp, int Kin, int ntiles,
    const float* __restrict__ bias,
    float* __restrict__ Cf, long ldcf,
    unsigned short* __restrict__ Cb, long ldcb,
    long Mrows, int Nact, int relu)
{
    int lane = threadIdx.x & 31;
    int w    = threadIdx.x >> 5;
    int cg   = w & 1;                    // col group (2 x 16-col tiles each)
    int rg   = w >> 1;                   // row group (32 rows each)
    int nt0  = blockIdx.y * 4 + cg * 2;  // first 16-col tile
    if (nt0 >= ntiles) return;
    const bool t1 = (nt0 + 1) < ntiles;
    long m0 = (long)blockIdx.x * 128 + (long)rg * 32;
    int  kh = lane >> 4;                 // K-half select (ISA 16-bit layout)

    const unsigned short* arow0 = A + (m0 + (lane & 15)) * (long)lda + kh * 8;
    const unsigned short* arow1 = arow0 + 16 * (long)lda;
    const int KT = Kin >> 5;
    const unsigned short* bcol0 = Bp + (long)nt0 * KT * 512 + (long)lane * 16;
    const unsigned short* bcol1 = bcol0 + (long)KT * 512;

    v8f z = {0.f, 0.f, 0.f, 0.f, 0.f, 0.f, 0.f, 0.f};
    v8f acc00 = z, acc01 = z, acc10 = z, acc11 = z;

    if (t1)
        gemm_core<true>(arow0, arow1, bcol0, bcol1, Kin, acc00, acc01, acc10, acc11);
    else
        gemm_core<false>(arow0, arow1, bcol0, bcol1, Kin, acc00, acc01, acc10, acc11);

    auto store_tile = [&](const v8f& acc, long mbase, int nt) {
        int col = nt * 16 + (lane & 15);
        if (col >= Nact) return;
        float bv = bias ? bias[col] : 0.f;
        long rbase = mbase + kh * 8;
#pragma unroll
        for (int i = 0; i < 8; ++i) {
            long r = rbase + i;
            if (r < Mrows) {
                float v = acc[i] + bv;
                if (relu) v = fmaxf(v, 0.f);
                if (Cf) Cf[r * ldcf + col] = v;
                if (Cb) Cb[r * ldcb + col] = f2bf(v);
            }
        }
    };
    store_tile(acc00, m0, nt0);
    store_tile(acc10, m0 + 16, nt0);
    if (t1) {
        store_tile(acc01, m0, nt0 + 1);
        store_tile(acc11, m0 + 16, nt0 + 1);
    }
}

// ---------------------------------------------------------------------------
// Feature concat + embedding gather -> bf16 [NP, 256]
// ---------------------------------------------------------------------------
__global__ __launch_bounds__(256) void encode_k(
    const int* __restrict__ ps, const int* __restrict__ ks,
    const float* __restrict__ xn,
    const float* __restrict__ embp, const float* __restrict__ embk,
    unsigned short* __restrict__ zb, int N, long NP)
{
    long tid = (long)blockIdx.x * 256 + threadIdx.x;
    if (tid >= NP * 256) return;
    long n = tid >> 8; int c = (int)(tid & 255);
    float v = 0.f;
    if (n < N) {
        if (c < 128)       v = xn[n * 128 + c];
        else if (c < 192)  v = embp[(long)ps[n] * 64 + (c - 128)];
        else               v = embk[(long)ks[n] * 64 + (c - 192)];
    }
    zb[tid] = f2bf(v);
}

// ---------------------------------------------------------------------------
// LayerNorm: one wave32 per row, __shfl_xor reduction; optional per-task
// gamma/beta via tdiv (param offset = (row/tdiv)*D).  In-place f32 OK.
// ---------------------------------------------------------------------------
__global__ __launch_bounds__(256) void layernorm_k(
    const float* __restrict__ X, int D,
    const float* __restrict__ g, const float* __restrict__ b,
    float* __restrict__ Of, unsigned short* __restrict__ Ob,
    long rows, long tdiv)
{
    long row = (long)blockIdx.x * 8 + (threadIdx.x >> 5);
    if (row >= rows) return;
    int lane = threadIdx.x & 31;
    const float* x = X + row * D;
    long po = tdiv ? (row / tdiv) * D : 0;
    const float* gg = g + po;
    const float* bb = b + po;
    float s = 0.f, s2 = 0.f;
    for (int c = lane; c < D; c += 32) { float v = x[c]; s += v; s2 += v * v; }
    for (int m = 16; m >= 1; m >>= 1) {
        s  += __shfl_xor(s,  m, 32);
        s2 += __shfl_xor(s2, m, 32);
    }
    float inv  = 1.0f / (float)D;
    float mean = s * inv;
    float var  = s2 * inv - mean * mean;
    float rs   = rsqrtf(var + 1e-5f);
    for (int c = lane; c < D; c += 32) {
        float y = (x[c] - mean) * rs * gg[c] + bb[c];
        if (Of) Of[row * (long)D + c] = y;
        if (Ob) Ob[row * (long)D + c] = f2bf(y);
    }
}

// ---------------------------------------------------------------------------
// Segment-mean scatter pieces (GNN message passing), D = 512 fixed.
// ---------------------------------------------------------------------------
__global__ __launch_bounds__(256) void scat_count_k(
    const int* __restrict__ dst, float* __restrict__ cnt, int E)
{
    int e = blockIdx.x * 256 + threadIdx.x;
    if (e < E) atomicAdd(&cnt[dst[e]], 1.0f);
}

__global__ __launch_bounds__(256) void scat_add_k(
    const float* __restrict__ Hr, const int* __restrict__ src,
    const int* __restrict__ dst, int E, float* __restrict__ S)
{
    long tid = (long)blockIdx.x * 256 + threadIdx.x;
    if (tid >= (long)E * 128) return;
    int e = (int)(tid >> 7);
    int c = (int)(tid & 127) << 2;
    long so = (long)src[e] * 512 + c;
    long dd = (long)dst[e] * 512 + c;
#pragma unroll
    for (int j = 0; j < 4; ++j) atomicAdd(&S[dd + j], Hr[so + j]);
}

__global__ __launch_bounds__(256) void seg_div_acc_k(
    const float* __restrict__ S, const float* __restrict__ cnt,
    float* __restrict__ M, long NP)
{
    long tid = (long)blockIdx.x * 256 + threadIdx.x;
    if (tid >= NP * 512) return;
    long n = tid >> 9;
    M[tid] += S[tid] / fmaxf(cnt[n], 1.0f);
}

__global__ __launch_bounds__(256) void gnn_combine_k(
    const float* __restrict__ Hself, const float* __restrict__ M,
    float* __restrict__ Xf, unsigned short* __restrict__ Xb, long total)
{
    long tid = (long)blockIdx.x * 256 + threadIdx.x;
    if (tid >= total) return;
    float v = fmaxf(Hself[tid] + M[tid], 0.f);
    Xf[tid] = v;
    Xb[tid] = f2bf(v);
}

// ---------------------------------------------------------------------------
// Onset-edge pooling
// ---------------------------------------------------------------------------
__global__ __launch_bounds__(256) void pool_count_k(
    const int* __restrict__ src, const int* __restrict__ dst,
    float* __restrict__ cnt, int E, int bs)
{
    int e = blockIdx.x * 256 + threadIdx.x;
    if (e >= E) return;
    int s = src[e], d = dst[e];
    if (s < bs && d < bs && s != d) atomicAdd(&cnt[s], 1.0f);
}

__global__ __launch_bounds__(256) void pool_scat_k(
    const float* __restrict__ X, const int* __restrict__ src,
    const int* __restrict__ dst, int E, int bs, float* __restrict__ S)
{
    long tid = (long)blockIdx.x * 256 + threadIdx.x;
    if (tid >= (long)E * 128) return;
    int e = (int)(tid >> 7);
    int s = src[e], d = dst[e];
    if (!(s < bs && d < bs && s != d)) return;
    int c = (int)(tid & 127) << 2;
    long so = (long)s * 512 + c;
    long dd = (long)d * 512 + c;
#pragma unroll
    for (int j = 0; j < 4; ++j) atomicAdd(&S[so + j], X[dd + j]);
}

__global__ __launch_bounds__(256) void build_xcat_k(
    const float* __restrict__ X, const float* __restrict__ S,
    const float* __restrict__ cnt, float* __restrict__ Xcat, long NP)
{
    long tid = (long)blockIdx.x * 256 + threadIdx.x;
    if (tid >= NP * 512) return;
    long n = tid >> 9; int c = (int)(tid & 511);
    float x = X[tid];
    Xcat[n * 1024 + c]       = x;
    Xcat[n * 1024 + 512 + c] = (x + S[tid]) / fmaxf(cnt[n], 1.0f);
}

__global__ __launch_bounds__(256) void add_res_k(
    float* __restrict__ O, const float* __restrict__ P, long total)
{
    long tid = (long)blockIdx.x * 256 + threadIdx.x;
    if (tid < total) O[tid] += P[tid];
}

// ---------------------------------------------------------------------------
// Cross-task attention over T=4 tokens, HEADS=4, DH=64.
// One thread per (node, head, t).  qkv rows live at (t*NP + n) * 768,
// [q | k | v] thirds of 256 each; o written bf16 at (t*NP + n)*256 + h*64.
// ---------------------------------------------------------------------------
__global__ __launch_bounds__(256) void attention4_k(
    const unsigned short* __restrict__ qkv, unsigned short* __restrict__ Ob,
    long NP, int N)
{
    long tid = (long)blockIdx.x * 256 + threadIdx.x;
    if (tid >= (long)N * 16) return;
    long n = tid >> 4;
    int  h = (int)((tid >> 2) & 3);
    int  t = (int)(tid & 3);

    float q[64];
    {
        const unsigned short* qp = qkv + ((long)t * NP + n) * 768 + h * 64;
#pragma unroll
        for (int d = 0; d < 64; ++d) q[d] = bf2f(qp[d]);
    }
    float sc[4];
#pragma unroll
    for (int s = 0; s < 4; ++s) {
        const unsigned short* kp = qkv + ((long)s * NP + n) * 768 + 256 + h * 64;
        float acc = 0.f;
#pragma unroll
        for (int d = 0; d < 64; ++d) acc += q[d] * bf2f(kp[d]);
        sc[s] = acc * 0.125f;   // / sqrt(64)
    }
    float mx = fmaxf(fmaxf(sc[0], sc[1]), fmaxf(sc[2], sc[3]));
    float ssum = 0.f;
#pragma unroll
    for (int s = 0; s < 4; ++s) { sc[s] = __expf(sc[s] - mx); ssum += sc[s]; }
    float rinv = 1.0f / ssum;

    float o[64];
#pragma unroll
    for (int d = 0; d < 64; ++d) o[d] = 0.f;
#pragma unroll
    for (int s = 0; s < 4; ++s) {
        const unsigned short* vp = qkv + ((long)s * NP + n) * 768 + 512 + h * 64;
        float a = sc[s] * rinv;
#pragma unroll
        for (int d = 0; d < 64; ++d) o[d] += a * bf2f(vp[d]);
    }
    unsigned short* op = Ob + ((long)t * NP + n) * 256 + h * 64;
#pragma unroll
    for (int d = 0; d < 64; ++d) op[d] = f2bf(o[d]);
}

// ---------------------------------------------------------------------------
// Host-side launch
// ---------------------------------------------------------------------------
extern "C" void kernel_launch(void* const* d_in, const int* in_sizes, int n_in,
                              void* d_out, int out_size, void* d_ws, size_t ws_size,
                              hipStream_t stream)
{
    (void)n_in; (void)out_size; (void)ws_size;

    const int*   ps = (const int*)  d_in[0];
    const int*   ks = (const int*)  d_in[1];
    const float* xn = (const float*)d_in[2];
    const int*   eo = (const int*)  d_in[3];   // [2,E] edge_onset
    const int*   ec = (const int*)  d_in[4];   // [2,E] edge_consecutive
    const int N  = in_sizes[0];
    const int E  = in_sizes[3] / 2;

    // params flattened in dict insertion order starting at d_in[6]
    const float* emb_pitch = (const float*)d_in[6];
    const float* emb_key   = (const float*)d_in[7];
    const float* proj_W1   = (const float*)d_in[8];
    const float* proj_b1   = (const float*)d_in[9];
    const float* proj_lng  = (const float*)d_in[10];
    const float* proj_lnb  = (const float*)d_in[11];
    const float* proj_W2   = (const float*)d_in[12];
    const float* proj_b2   = (const float*)d_in[13];
    const float* enc_Wself[2] = { (const float*)d_in[14], (const float*)d_in[18] };
    const float* enc_b[2]     = { (const float*)d_in[15], (const float*)d_in[19] };
    const float* enc_Wrel[2][2] = {
        { (const float*)d_in[16], (const float*)d_in[17] },
        { (const float*)d_in[20], (const float*)d_in[21] } };
    const float* pe_ln0g = (const float*)d_in[22];
    const float* pe_ln0b = (const float*)d_in[23];
    const float* pe_W1   = (const float*)d_in[24];
    const float* pe_b1   = (const float*)d_in[25];
    const float* pe_ln1g = (const float*)d_in[26];
    const float* pe_ln1b = (const float*)d_in[27];
    const float* pe_W2   = (const float*)d_in[28];
    const float* pe_b2   = (const float*)d_in[29];
    const float* pe_ln2g = (const float*)d_in[30];
    const float* pe_ln2b = (const float*)d_in[31];
    const float* pe_W3   = (const float*)d_in[32];
    const float* pe_b3   = (const float*)d_in[33];
    const float* clf_W1  = (const float*)d_in[34];  // [4,512,256]
    const float* clf_b1  = (const float*)d_in[35];  // [4,256]
    const float* clf_lng = (const float*)d_in[36];
    const float* clf_lnb = (const float*)d_in[37];
    const float* clf_W2  = (const float*)d_in[38];  // [4,256,38]
    const float* clf_b2  = (const float*)d_in[39];  // [4,38]
    const float* tp_W    = (const float*)d_in[40];  // [4,38,256]
    const float* tp_b    = (const float*)d_in[41];  // [4,256]
    const float* tp_lng  = (const float*)d_in[42];
    const float* tp_lnb  = (const float*)d_in[43];
    const float* fu_W    = (const float*)d_in[44];  // [4,256,38]
    const float* fu_b    = (const float*)d_in[45];  // [4,38]
    const float* attn_Wqkv = (const float*)d_in[46]; // [768,256]
    const float* attn_bqkv = (const float*)d_in[47];
    const float* attn_Wo   = (const float*)d_in[48]; // [256,256]
    const float* attn_bo   = (const float*)d_in[49];
    const float* attn_lng  = (const float*)d_in[50];
    const float* attn_lnb  = (const float*)d_in[51];

    const long NP  = ((long)N + 127) & ~127L;   // 128-row GEMM tiles
    const long TNP = 4 * NP;
    const int  T = 4, NC = 38;

    // ---- workspace carve-out ----
    char* base = (char*)d_ws;
    size_t off = 0;
    auto alloc = [&](size_t bytes) -> void* {
        off = (off + 255) & ~(size_t)255;
        void* p = base + off;
        off += bytes;
        return p;
    };
    auto allocU = [&](long elems) { return (unsigned short*)alloc((size_t)elems * 2); };
    auto allocF = [&](long elems) { return (float*)alloc((size_t)elems * 4); };

    // packed weights (bf16)
    unsigned short* pk_proj_W1 = allocU(256L * 512);
    unsigned short* pk_proj_W2 = allocU(512L * 512);
    unsigned short* pk_enc_self[2]; unsigned short* pk_enc_rel[2][2];
    for (int l = 0; l < 2; ++l) {
        pk_enc_self[l] = allocU(512L * 512);
        for (int r = 0; r < 2; ++r) pk_enc_rel[l][r] = allocU(512L * 512);
    }
    unsigned short* pk_pe_W1 = allocU(1024L * 512);
    unsigned short* pk_pe_W2 = allocU(512L * 512);
    unsigned short* pk_pe_W3 = allocU(512L * 512);
    unsigned short* pk_clf_W1[4]; unsigned short* pk_clf_W2[4];
    unsigned short* pk_tp_W[4];   unsigned short* pk_fu_W[4];
    for (int t = 0; t < 4; ++t) {
        pk_clf_W1[t] = allocU(512L * 256);
        pk_clf_W2[t] = allocU(256L * 48);
        pk_tp_W[t]   = allocU(64L * 256);
        pk_fu_W[t]   = allocU(256L * 48);
    }
    unsigned short* pk_qkv = allocU(256L * 768);
    unsigned short* pk_wo  = allocU(256L * 256);

    // activations
    unsigned short* zb   = allocU(NP * 256);
    float* F1   = allocF(NP * 512);   // h / x (f32)
    float* F2   = allocF(NP * 512);   // hr / hself / pe temps
    float* F3   = allocF(NP * 512);   // scatter sum S
    float* Macc = allocF(NP * 512);   // msg accumulator
    float* cnt  = allocF(NP);
    unsigned short* Hb  = allocU(NP * 512);
    unsigned short* Hb2 = allocU(NP * 512);
    float*          Xcat = allocF(NP * 1024);
    unsigned short* Xcb  = allocU(NP * 1024);
    // task-major buffers [T][NP][*]
    float*          Tf   = allocF(TNP * 256);   // r1 f32
    unsigned short* Tb   = allocU(TNP * 256);   // r1_b -> o_b -> enh_b (reused)
    float*          Pf   = allocF(TNP * 256);   // pr (LN'd, f32 for residual)
    unsigned short* Pb   = allocU(TNP * 256);   // pr bf16
    unsigned short* rawb = allocU(TNP * 64);    // raw logits bf16, K-padded to 64
    unsigned short* qkvb = allocU(TNP * 768);
    float*          Of   = allocF(TNP * 256);   // attn out -> pr+o

    float* outp = (float*)d_out;

    auto packw = [&](const float* W, unsigned short* P, int Ka, int Na,
                     int Kp, int Np, int tr) {
        long total = (long)(Np >> 4) * (Kp >> 5) * 512;
        pack_weight_k<<<dim3((unsigned)((total + 255) / 256)), 256, 0, stream>>>(
            W, P, Ka, Na, Kp, Np, tr);
    };
    auto gemm = [&](const unsigned short* A, int lda, const unsigned short* Bp,
                    int Kin, int Npad, const float* bias,
                    float* Cf, long ldcf, unsigned short* Cb, long ldcb,
                    long Mpad, long Mrows, int Nact, int relu) {
        dim3 grid((unsigned)(Mpad / 128), (unsigned)((Npad + 63) / 64));
        gemm_bf16_wmma<<<grid, 256, 0, stream>>>(A, lda, Bp, Kin, Npad / 16,
                                                 bias, Cf, ldcf, Cb, ldcb,
                                                 Mrows, Nact, relu);
    };
    auto ln = [&](const float* X, int D, const float* g, const float* b,
                  float* Ofp, unsigned short* Obp, long rows, long tdiv) {
        layernorm_k<<<dim3((unsigned)((rows + 7) / 8)), 256, 0, stream>>>(
            X, D, g, b, Ofp, Obp, rows, tdiv);
    };

    // ---- 0: pack all weights ----
    packw(proj_W1, pk_proj_W1, 256, 512, 256, 512, 0);
    packw(proj_W2, pk_proj_W2, 512, 512, 512, 512, 0);
    for (int l = 0; l < 2; ++l) {
        packw(enc_Wself[l], pk_enc_self[l], 512, 512, 512, 512, 0);
        for (int r = 0; r < 2; ++r)
            packw(enc_Wrel[l][r], pk_enc_rel[l][r], 512, 512, 512, 512, 0);
    }
    packw(pe_W1, pk_pe_W1, 1024, 512, 1024, 512, 0);
    packw(pe_W2, pk_pe_W2, 512, 512, 512, 512, 0);
    packw(pe_W3, pk_pe_W3, 512, 512, 512, 512, 0);
    for (int t = 0; t < 4; ++t) {
        packw(clf_W1 + (long)t * 512 * 256, pk_clf_W1[t], 512, 256, 512, 256, 0);
        packw(clf_W2 + (long)t * 256 * NC, pk_clf_W2[t], 256, NC, 256, 48, 0);
        packw(tp_W   + (long)t * NC * 256, pk_tp_W[t],   NC, 256, 64, 256, 0);
        packw(fu_W   + (long)t * 256 * NC, pk_fu_W[t],   256, NC, 256, 48, 0);
    }
    packw(attn_Wqkv, pk_qkv, 256, 768, 256, 768, 1);  // x @ W.T -> transposed
    packw(attn_Wo,   pk_wo,  256, 256, 256, 256, 1);

    // ---- 1: encode ----
    encode_k<<<dim3((unsigned)((NP * 256 + 255) / 256)), 256, 0, stream>>>(
        ps, ks, xn, emb_pitch, emb_key, zb, N, NP);
    gemm(zb, 256, pk_proj_W1, 256, 512, proj_b1, F1, 512, nullptr, 0, NP, NP, 512, 1);
    ln(F1, 512, proj_lng, proj_lnb, nullptr, Hb, NP, 0);
    gemm(Hb, 512, pk_proj_W2, 512, 512, proj_b2, F1, 512, Hb2, 512, NP, NP, 512, 0);
    // h now in Hb2 (bf16) / F1 (f32)

    // ---- 2: GNN layers ----
    const int* rel_src[2] = { eo,     ec     };
    const int* rel_dst[2] = { eo + E, ec + E };
    unsigned short* hin[2]  = { Hb2, Hb  };
    unsigned short* hout[2] = { Hb,  Hb2 };
    for (int l = 0; l < 2; ++l) {
        hipMemsetAsync(Macc, 0, (size_t)NP * 512 * 4, stream);
        for (int r = 0; r < 2; ++r) {
            gemm(hin[l], 512, pk_enc_rel[l][r], 512, 512, nullptr,
                 F2, 512, nullptr, 0, NP, NP, 512, 0);
            hipMemsetAsync(F3,  0, (size_t)NP * 512 * 4, stream);
            hipMemsetAsync(cnt, 0, (size_t)NP * 4, stream);
            scat_count_k<<<dim3((E + 255) / 256), 256, 0, stream>>>(rel_dst[r], cnt, E);
            scat_add_k<<<dim3((unsigned)(((long)E * 128 + 255) / 256)), 256, 0, stream>>>(
                F2, rel_src[r], rel_dst[r], E, F3);
            seg_div_acc_k<<<dim3((unsigned)((NP * 512 + 255) / 256)), 256, 0, stream>>>(
                F3, cnt, Macc, NP);
        }
        gemm(hin[l], 512, pk_enc_self[l], 512, 512, enc_b[l],
             F2, 512, nullptr, 0, NP, NP, 512, 0);
        gnn_combine_k<<<dim3((unsigned)((NP * 512 + 255) / 256)), 256, 0, stream>>>(
            F2, Macc, F1, hout[l], NP * 512);
    }
    // x: f32 in F1, bf16 in Hb2

    // ---- 3: onset pooling + concat + project_enc ----
    hipMemsetAsync(F3,  0, (size_t)NP * 512 * 4, stream);
    hipMemsetAsync(cnt, 0, (size_t)NP * 4, stream);
    pool_count_k<<<dim3((E + 255) / 256), 256, 0, stream>>>(eo, eo + E, cnt, E, N);
    pool_scat_k<<<dim3((unsigned)(((long)E * 128 + 255) / 256)), 256, 0, stream>>>(
        F1, eo, eo + E, E, N, F3);
    build_xcat_k<<<dim3((unsigned)((NP * 512 + 255) / 256)), 256, 0, stream>>>(
        F1, F3, cnt, Xcat, NP);
    ln(Xcat, 1024, pe_ln0g, pe_ln0b, nullptr, Xcb, NP, 0);
    gemm(Xcb, 1024, pk_pe_W1, 1024, 512, pe_b1, F2, 512, nullptr, 0, NP, NP, 512, 1);
    ln(F2, 512, pe_ln1g, pe_ln1b, nullptr, Hb, NP, 0);
    gemm(Hb, 512, pk_pe_W2, 512, 512, pe_b2, F2, 512, nullptr, 0, NP, NP, 512, 1);
    ln(F2, 512, pe_ln2g, pe_ln2b, nullptr, Hb2, NP, 0);
    gemm(Hb2, 512, pk_pe_W3, 512, 512, pe_b3, nullptr, 0, Hb, 512, NP, NP, 512, 0);
    // projected x (bf16) in Hb

    // ---- 4: per-task classification heads ----
    for (int t = 0; t < 4; ++t)
        gemm(Hb, 512, pk_clf_W1[t], 512, 256, clf_b1 + t * 256,
             Tf + (long)t * NP * 256, 256, nullptr, 0, NP, NP, 256, 1);
    ln(Tf, 256, clf_lng, clf_lnb, nullptr, Tb, TNP, NP);
    hipMemsetAsync(rawb, 0, (size_t)TNP * 64 * 2, stream);
    for (int t = 0; t < 4; ++t)
        gemm(Tb + (long)t * NP * 256, 256, pk_clf_W2[t], 256, 48, clf_b2 + t * NC,
             nullptr, 0, rawb + (long)t * NP * 64, 64, NP, NP, NC, 0);

    // ---- 5: logit fusion ----
    for (int t = 0; t < 4; ++t)
        gemm(rawb + (long)t * NP * 64, 64, pk_tp_W[t], 64, 256, tp_b + t * 256,
             Pf + (long)t * NP * 256, 256, nullptr, 0, NP, NP, 256, 1);
    ln(Pf, 256, tp_lng, tp_lnb, Pf, Pb, TNP, NP);   // in-place f32 + bf16
    gemm(Pb, 256, pk_qkv, 256, 768, attn_bqkv,
         nullptr, 0, qkvb, 768, TNP, TNP, 768, 0);
    attention4_k<<<dim3((unsigned)(((long)N * 16 + 255) / 256)), 256, 0, stream>>>(
        qkvb, Tb, NP, N);                            // Tb reused as o (bf16)
    gemm(Tb, 256, pk_wo, 256, 256, attn_bo, Of, 256, nullptr, 0, TNP, TNP, 256, 0);
    add_res_k<<<dim3((unsigned)((TNP * 256 + 255) / 256)), 256, 0, stream>>>(
        Of, Pf, TNP * 256);
    ln(Of, 256, attn_lng, attn_lnb, nullptr, Tb, TNP, 0);  // enh (bf16) in Tb
    for (int t = 0; t < 4; ++t)
        gemm(Tb + (long)t * NP * 256, 256, pk_fu_W[t], 256, 48, fu_b + t * NC,
             outp + (long)t * NC, (long)T * NC, nullptr, 0, NP, N, NC, 0);
}
